// AttentionModel_47648367182457
// MI455X (gfx1250) — matrix operand order repaired and checked by reading, not compiled, not run on hardware
//
#include <hip/hip_runtime.h>
#include <math.h>
#include <stdint.h>

// ---------------------------------------------------------------------------
// AttentionModel pointer-network log_p, fused for MI455X (gfx1250, wave32).
//
//  * one workgroup (256 thr = 8 waves) per batch; wave w == head h
//  * pass0: mean(E[b]) -> query (VALU, coalesced)
//  * pass1: for 16 chunks of 64 keys:
//       - chunk ch+1 raw fp32 is DMA'd global->LDS with
//         GLOBAL_LOAD_ASYNC_TO_LDS_B128 (ASYNCcnt) while chunk ch computes
//       - bf16 hi/lo split from LDS staging -> aHi/aLo (convert once/elem)
//       - K/V = chunk @ W_node[:,0:512] via v_wmma_f32_16x16x32_bf16
//         (3 WMMAs per tile: hi*hi + hi*lo + lo*hi ~ fp32 accuracy)
//       - each wave owns exactly its head's 64 columns; flash-style online
//         softmax entirely in registers (wave32 shfl butterflies)
//  * glimpse = heads @ W_out ; w2 = W_node[:,512:768] @ glimpse
//  * pass2: logits[n] = E[b,n,:]·w2 / 16 -> tanh clip -> mask -> log_softmax
//  kvl is never materialized to HBM.
// ---------------------------------------------------------------------------

#define BB 256
#define NN 1024
#define DD 256
#define HH 8
#define NC 64                 // keys per chunk
#define NCHUNK (NN / NC)      // 16
#define KSTEPS (DD / 32)      // 8

typedef __attribute__((ext_vector_type(16))) __bf16 v16bf;
typedef __attribute__((ext_vector_type(8)))  float  v8f;

union Frag {
    unsigned u[8];
    uint4    q[2];
    v16bf    v;
};

__device__ __forceinline__ unsigned short f2bf(float x) {
    unsigned b = __float_as_uint(x);
    return (unsigned short)((b + 0x7fffu + ((b >> 16) & 1u)) >> 16);   // RNE
}
__device__ __forceinline__ float bf2f(unsigned short h) {
    return __uint_as_float(((unsigned)h) << 16);
}
// split two fp32 into packed bf16 hi and bf16 residual lo (lo16 = even k)
__device__ __forceinline__ void split2(float x0, float x1, unsigned& hi, unsigned& lo) {
    unsigned short h0 = f2bf(x0), h1 = f2bf(x1);
    float r0 = x0 - bf2f(h0), r1 = x1 - bf2f(h1);
    unsigned short l0 = f2bf(r0), l1 = f2bf(r1);
    hi = (unsigned)h0 | ((unsigned)h1 << 16);
    lo = (unsigned)l0 | ((unsigned)l1 << 16);
}

// CDNA5 async global->LDS DMA (tracked by ASYNCcnt). ldsOff = LDS byte addr.
__device__ __forceinline__ void async_cp16(unsigned ldsOff, const void* gaddr) {
    asm volatile("global_load_async_to_lds_b128 %0, %1, off"
                 :: "v"(ldsOff), "v"(gaddr) : "memory");
}
__device__ __forceinline__ void async_wait0() {
    asm volatile("s_wait_asynccnt 0x0" ::: "memory");
}

// ---- prep: W_node[:,0:512] -> transposed, bf16 hi/lo, K-pair packed --------
// Wt[n][kp] packs W_node[2kp][n] (lo16) and W_node[2kp+1][n] (hi16).
__global__ __launch_bounds__(256) void prep_wt(const float* __restrict__ Wn,
                                               unsigned* __restrict__ WtHi,
                                               unsigned* __restrict__ WtLo) {
    int idx = blockIdx.x * 256 + threadIdx.x;      // 512*128 exactly
    int n = idx >> 7, kp = idx & 127;
    float x0 = Wn[(2 * kp) * 768 + n];
    float x1 = Wn[(2 * kp + 1) * 768 + n];
    unsigned hi, lo;
    split2(x0, x1, hi, lo);
    WtHi[idx] = hi;
    WtLo[idx] = lo;
}

// ---- main fused kernel -----------------------------------------------------
__global__ __launch_bounds__(256, 1) void attn_fused(
    const float* __restrict__ E,   const float* __restrict__ SC,
    const unsigned char* __restrict__ MK,
    const float* __restrict__ Wn,  const float* __restrict__ Wf,
    const float* __restrict__ Ws,  const float* __restrict__ Wo,
    const unsigned* __restrict__ WtHi, const unsigned* __restrict__ WtLo,
    float* __restrict__ out) {

    __shared__ __align__(16) float eStage[NC * DD];  // raw fp32 chunk (64 KB)
    __shared__ unsigned aHi[NC * 128];   // A chunk, bf16-hi pairs  (32 KB)
    __shared__ unsigned aLo[NC * 128];   // A chunk, bf16-lo pairs  (32 KB)
    __shared__ float meanS[DD];
    __shared__ float qS[DD];
    __shared__ float headsS[DD];
    __shared__ float glimS[DD];
    __shared__ float w2S[DD];
    __shared__ float logitS[NN];
    __shared__ float redS[HH];

    const int t    = threadIdx.x;
    const int b    = blockIdx.x;
    const int w    = t >> 5;        // wave id == head id
    const int lane = t & 31;
    const int h2   = lane >> 4;     // half-wave
    const int n16  = lane & 15;

    const float* Eb = E + (size_t)b * (NN * DD);
    const unsigned char* mb = MK + b * NN;

    // LDS byte address of the async staging buffer (addr[31:0] of generic ptr)
    const unsigned eBase = (unsigned)(uintptr_t)&eStage[0];

    // ---------------- pass 0: graph mean -> query ----------------
    {
        float s = 0.f;
        #pragma unroll 4
        for (int n = 0; n < NN; ++n) s += Eb[n * DD + t];
        meanS[t] = s * (1.0f / NN);
    }
    __syncthreads();
    {
        float qv = 0.f;
        #pragma unroll 4
        for (int i = 0; i < DD; ++i) qv += meanS[i] * Wf[i * DD + t];
        const float* scb = SC + b * (2 * DD + 1);
        #pragma unroll 4
        for (int j = 0; j < 2 * DD + 1; ++j) qv += scb[j] * Ws[j * DD + t];
        qS[t] = qv;
    }
    __syncthreads();

    const float q0 = qS[w * 32 + n16];        // this head's query, d = jt*16+n16
    const float q1 = qS[w * 32 + 16 + n16];
    const float rsd = 0.17677669529663687f;   // 1/sqrt(32)

    float Mrun = -1e30f, Lrun = 0.f;
    float vacc0 = 0.f, vacc1 = 0.f;           // softmax numerator, per-half partial

    // kick off async DMA of chunk 0 (each thread moves 4 x 16B, 16 KB total)
    {
        const float* src = Eb;                // chunk 0 is contiguous
        #pragma unroll
        for (int i = 0; i < 4; ++i) {
            int f4 = i * 256 + t;
            async_cp16(eBase + f4 * 16, src + f4 * 4);
        }
    }

    for (int ch = 0; ch < NCHUNK; ++ch) {
        // wait for this chunk's DMA, make it visible to all waves
        async_wait0();
        __syncthreads();

        // bf16 hi/lo split: LDS fp32 staging -> aHi/aLo (convert once/elem)
        #pragma unroll 4
        for (int i = 0; i < (NC * 128) / 256; ++i) {
            int idx = i * 256 + t;
            const float2 e2 = ((const float2*)eStage)[idx];
            unsigned hi, lo;
            split2(e2.x, e2.y, hi, lo);
            aHi[idx] = hi;
            aLo[idx] = lo;
        }
        __syncthreads();                      // aHi/aLo ready; eStage consumed

        // prefetch chunk ch+1 via async DMA; overlaps with WMMA below
        if (ch + 1 < NCHUNK) {
            const float* src = Eb + (ch + 1) * (NC * DD);
            #pragma unroll
            for (int i = 0; i < 4; ++i) {
                int f4 = i * 256 + t;
                async_cp16(eBase + f4 * 16, src + f4 * 4);
            }
        }

        // K/V accumulators: 4 M-tiles x (2 key + 2 val) N-tiles, fp32
        v8f accK[4][2], accV[4][2];
        const v8f vzero = {0.f, 0.f, 0.f, 0.f, 0.f, 0.f, 0.f, 0.f};
        #pragma unroll
        for (int mt = 0; mt < 4; ++mt)
            #pragma unroll
            for (int jt = 0; jt < 2; ++jt) { accK[mt][jt] = vzero; accV[mt][jt] = vzero; }

        #pragma unroll 1
        for (int ks = 0; ks < KSTEPS; ++ks) {
            // A fragments (16x32 bf16): vgpr v -> kpair (v<4? v : v+4) + 4*h2
            Frag aH[4], aL[4];
            #pragma unroll
            for (int mt = 0; mt < 4; ++mt) {
                int base = (mt * 16 + n16) * 128 + ks * 16 + 4 * h2;
                aH[mt].q[0] = *(const uint4*)&aHi[base];
                aH[mt].q[1] = *(const uint4*)&aHi[base + 8];
                aL[mt].q[0] = *(const uint4*)&aLo[base];
                aL[mt].q[1] = *(const uint4*)&aLo[base + 8];
            }
            #pragma unroll
            for (int ct = 0; ct < 4; ++ct) {
                // B fragments (32x16 bf16): vgpr j -> kpair (j<4? j : j+4) + 4*h2
                int cb = (ct < 2) ? (w * 32 + ct * 16) : (256 + w * 32 + (ct - 2) * 16);
                int boff = (cb + n16) * 128 + ks * 16 + 4 * h2;
                Frag bH, bL;
                bH.q[0] = *(const uint4*)&WtHi[boff];
                bH.q[1] = *(const uint4*)&WtHi[boff + 8];
                bL.q[0] = *(const uint4*)&WtLo[boff];
                bL.q[1] = *(const uint4*)&WtLo[boff + 8];
                #pragma unroll
                for (int mt = 0; mt < 4; ++mt) {
                    v8f c = (ct < 2) ? accK[mt][ct] : accV[mt][ct - 2];
                    c = __builtin_amdgcn_wmma_f32_16x16x32_bf16(false, aH[mt].v, false, bH.v,
                                                                (short)0, c, false, false);
                    c = __builtin_amdgcn_wmma_f32_16x16x32_bf16(false, aH[mt].v, false, bL.v,
                                                                (short)0, c, false, false);
                    c = __builtin_amdgcn_wmma_f32_16x16x32_bf16(false, aL[mt].v, false, bH.v,
                                                                (short)0, c, false, false);
                    if (ct < 2) accK[mt][ct] = c; else accV[mt][ct - 2] = c;
                }
            }
        }
        __syncthreads();   // all waves done reading aHi/aLo

        // ---- compat[n] = q . K[n] for the 64 chunk keys (register-only) ----
        // C layout: vgpr r, lane -> key row n = mt*16 + r + 8*h2, d = lane%16.
        float cv[2];  cv[0] = -1e30f;  cv[1] = -1e30f;   // cv[c]: n = c*32+lane
        #pragma unroll
        for (int mt = 0; mt < 4; ++mt) {
            #pragma unroll
            for (int r = 0; r < 8; ++r) {
                float tt = q0 * accK[mt][0][r] + q1 * accK[mt][1][r];
                tt += __shfl_xor(tt, 1);  tt += __shfl_xor(tt, 2);
                tt += __shfl_xor(tt, 4);  tt += __shfl_xor(tt, 8);
                float sx = __shfl_xor(tt, 16);
                float s0 = (lane < 16) ? tt : sx;   // half0 sum -> key mt*16+r
                float s1 = (lane < 16) ? sx : tt;   // half1 sum -> key mt*16+r+8
                int nl = mt * 16 + r, nh = nl + 8;
                if (lane == (nl & 31)) cv[nl >> 5] = s0;
                if (lane == (nh & 31)) cv[nh >> 5] = s1;
            }
        }
        cv[0] *= rsd;  cv[1] *= rsd;
        if (mb[ch * NC + lane])      cv[0] = -1e30f;
        if (mb[ch * NC + 32 + lane]) cv[1] = -1e30f;

        // ---- online softmax update ----
        float cm = fmaxf(cv[0], cv[1]);
        cm = fmaxf(cm, __shfl_xor(cm, 1));  cm = fmaxf(cm, __shfl_xor(cm, 2));
        cm = fmaxf(cm, __shfl_xor(cm, 4));  cm = fmaxf(cm, __shfl_xor(cm, 8));
        cm = fmaxf(cm, __shfl_xor(cm, 16));
        float Mn = fmaxf(Mrun, cm);
        float scale = __expf(Mrun - Mn);
        float p0 = __expf(cv[0] - Mn);
        float p1 = __expf(cv[1] - Mn);
        float ps = p0 + p1;
        ps += __shfl_xor(ps, 1);  ps += __shfl_xor(ps, 2);
        ps += __shfl_xor(ps, 4);  ps += __shfl_xor(ps, 8);
        ps += __shfl_xor(ps, 16);
        Lrun = Lrun * scale + ps;
        vacc0 *= scale;  vacc1 *= scale;
        Mrun = Mn;

        // ---- numerator += p[n] * V[n, d]  (V is layout-matched in accV) ----
        #pragma unroll
        for (int mt = 0; mt < 4; ++mt) {
            #pragma unroll
            for (int r = 0; r < 8; ++r) {
                int nidx = mt * 16 + r + 8 * h2;        // this half's key row
                float pn = (mt < 2) ? __shfl(p0, nidx & 31) : __shfl(p1, nidx & 31);
                vacc0 += pn * accV[mt][0][r];
                vacc1 += pn * accV[mt][1][r];
            }
        }
    }

    // combine the two half-wave partials, normalize, publish heads
    vacc0 += __shfl_xor(vacc0, 16);
    vacc1 += __shfl_xor(vacc1, 16);
    float invL = 1.0f / Lrun;
    if (lane < 16) {
        headsS[w * 32 + n16]      = vacc0 * invL;
        headsS[w * 32 + 16 + n16] = vacc1 * invL;
    }
    __syncthreads();

    // glimpse = heads @ W_out
    {
        float g = 0.f;
        #pragma unroll 4
        for (int i = 0; i < DD; ++i) g += headsS[i] * Wo[i * DD + t];
        glimS[t] = g;
    }
    __syncthreads();

    // w2 = W_node[:,512:768] @ glimpse   (folds logit_key GEMM away)
    {
        float s = 0.f;
        const float* wr = Wn + t * 768 + 512;
        #pragma unroll 4
        for (int d = 0; d < DD; ++d) s += wr[d] * glimS[d];
        w2S[t] = s;
    }
    __syncthreads();

    // ---- pass 2: logits[n] = E[b,n,:].w2 / 16 -> tanh clip -> mask ----
    {
        const float4* w2v = (const float4*)w2S;
        #pragma unroll
        for (int c = 0; c < 4; ++c) {
            int n = c * 256 + t;
            const float4* er = (const float4*)(Eb + n * DD);
            float s = 0.f;
            #pragma unroll 4
            for (int i = 0; i < 64; ++i) {
                float4 ev = er[i];
                float4 wv = w2v[i];
                s += ev.x * wv.x + ev.y * wv.y + ev.z * wv.z + ev.w * wv.w;
            }
            s = 10.0f * tanhf(s * 0.0625f);
            if (mb[n]) s = -__builtin_inff();
            logitS[n] = s;
        }
    }
    __syncthreads();

    // ---- log_softmax over N=1024 ----
    float lm = -__builtin_inff();
    #pragma unroll
    for (int c = 0; c < 4; ++c) lm = fmaxf(lm, logitS[c * 256 + t]);
    lm = fmaxf(lm, __shfl_xor(lm, 1));  lm = fmaxf(lm, __shfl_xor(lm, 2));
    lm = fmaxf(lm, __shfl_xor(lm, 4));  lm = fmaxf(lm, __shfl_xor(lm, 8));
    lm = fmaxf(lm, __shfl_xor(lm, 16));
    if (lane == 0) redS[w] = lm;
    __syncthreads();
    float gm = redS[0];
    #pragma unroll
    for (int i = 1; i < HH; ++i) gm = fmaxf(gm, redS[i]);

    float ls = 0.f;
    #pragma unroll
    for (int c = 0; c < 4; ++c) ls += __expf(logitS[c * 256 + t] - gm);
    ls += __shfl_xor(ls, 1);  ls += __shfl_xor(ls, 2);
    ls += __shfl_xor(ls, 4);  ls += __shfl_xor(ls, 8);
    ls += __shfl_xor(ls, 16);
    __syncthreads();                    // everyone done reading redS (max)
    if (lane == 0) redS[w] = ls;
    __syncthreads();
    float gs = 0.f;
    #pragma unroll
    for (int i = 0; i < HH; ++i) gs += redS[i];
    float lg = logf(gs);

    #pragma unroll
    for (int c = 0; c < 4; ++c) {
        int n = c * 256 + t;
        out[b * NN + n] = logitS[n] - gm - lg;
    }
}

// ---------------------------------------------------------------------------
extern "C" void kernel_launch(void* const* d_in, const int* in_sizes, int n_in,
                              void* d_out, int out_size, void* d_ws, size_t ws_size,
                              hipStream_t stream) {
    (void)in_sizes; (void)n_in; (void)out_size; (void)ws_size;
    const float* E  = (const float*)d_in[0];
    const float* SC = (const float*)d_in[1];
    const unsigned char* MK = (const unsigned char*)d_in[2];
    const float* Wn = (const float*)d_in[3];
    const float* Wf = (const float*)d_in[4];
    const float* Ws = (const float*)d_in[5];
    const float* Wo = (const float*)d_in[6];

    unsigned* WtHi = (unsigned*)d_ws;        // 512 cols x 128 kpairs
    unsigned* WtLo = WtHi + 512 * 128;       // 512 KB of workspace total

    prep_wt<<<256, 256, 0, stream>>>(Wn, WtHi, WtLo);
    attn_fused<<<BB, 256, 0, stream>>>(E, SC, MK, Wn, Wf, Ws, Wo, WtHi, WtLo,
                                       (float*)d_out);
}
